// SpikeTriggeredEventLayer_68702296867233
// MI455X (gfx1250) — matrix operand order repaired and checked by reading, not compiled, not run on hardware
//
#include <hip/hip_runtime.h>
#include <hip/hip_bf16.h>

typedef __attribute__((ext_vector_type(16))) _Float16 v16h;
typedef __attribute__((ext_vector_type(8)))  _Float16 v8h;
typedef __attribute__((ext_vector_type(8)))  float    v8f;
typedef __attribute__((ext_vector_type(4)))  float    f32x4;

#define B_   8
#define N_   8192
#define D_   256
#define KE_  64
#define HD_  64

// =====================================================================
// CDNA5 WMMA 16x16x32 f16 fragment layouts (CDNA5 ISA):
//  A (16x32): lane<16 -> M=lane, halves 0..7 -> K 0..7, halves 8..15 -> K 16..23;
//             lane>=16 adds +8 to K. => two contiguous 8-half (16B) runs per lane.
//  B (32x16): lane&15 = N, same K order as A (halves 0..7 -> K 0..7, 8..15 -> K 16..23, lane>=16 +8). Column-strided in
//             memory, so B matrices are PRE-PACKED in fragment order:
//             packed[(kt*16+nt)*32 + lane][h]  (one aligned 32B v16h per fragment).
//  C/D f32:   element r -> M = r + (lane>>4)*8, N = lane&15.
// =====================================================================

__device__ __forceinline__ v8f wmma_f16(v16h a, v16h b, v8f c) {
  return __builtin_amdgcn_wmma_f32_16x16x32_f16(false, a, false, b, (short)0, c, false, false);
}

__device__ __forceinline__ v16h load_a_f16(const _Float16* row, int k0, int lane) {
  int lh = (lane >> 4) << 3;
  v8h lo = *(const v8h*)(row + k0 + lh);        // 16B aligned
  v8h hi = *(const v8h*)(row + k0 + 16 + lh);
  v16h a;
  #pragma unroll
  for (int h = 0; h < 8; ++h) { a[h] = lo[h]; a[h + 8] = hi[h]; }
  return a;
}

__device__ __forceinline__ v16h load_a_f32(const float* row, int k0, int lane) {
  int lh = (lane >> 4) << 3;
  f32x4 x0 = *(const f32x4*)(row + k0 + lh);
  f32x4 x1 = *(const f32x4*)(row + k0 + lh + 4);
  f32x4 x2 = *(const f32x4*)(row + k0 + 16 + lh);
  f32x4 x3 = *(const f32x4*)(row + k0 + 16 + lh + 4);
  v16h a;
  #pragma unroll
  for (int t = 0; t < 4; ++t) {
    a[t]      = (_Float16)x0[t];
    a[4 + t]  = (_Float16)x1[t];
    a[8 + t]  = (_Float16)x2[t];
    a[12 + t] = (_Float16)x3[t];
  }
  return a;
}

// fragment-order output index for packing B-side matrices: element (k, n)
__device__ __forceinline__ size_t b_pack_idx(int k, int n) {
  int kt = k >> 5, r = k & 31;
  int lane = (((r >> 3) & 1) << 4) | (n & 15);
  int h = (r & 7) | ((r >> 4) << 3);
  int nt = n >> 4;
  return ((size_t)((kt * 16 + nt) * 32 + lane) << 4) + h;
}

// --- Hamilton-structured 256x256 weight -> packed B-fragment layout, f16 ---
__global__ void build_wt_kernel(const float* __restrict__ qr, const float* __restrict__ qi,
                                const float* __restrict__ qj, const float* __restrict__ qk,
                                _Float16* __restrict__ wt) {
  int gid = blockIdx.x * blockDim.x + threadIdx.x;   // 65536 elements
  int in  = gid >> 8;     // K dim (input)
  int out = gid & 255;    // N dim (output)
  int ob = out >> 6, ib = in >> 6, o = out & 63, ic = in & 63;
  // W blocks: [[r,-i,-j,-k],[i,r,-k,j],[j,k,r,-i],[k,-j,i,r]] ; Wt[in][out] = W[out][in]
  const int   srcT[16] = {0,1,2,3, 1,0,3,2, 2,3,0,1, 3,2,1,0};
  const float sgnT[16] = {1.f,-1.f,-1.f,-1.f, 1.f,1.f,-1.f,1.f,
                          1.f,1.f,1.f,-1.f,  1.f,-1.f,1.f,1.f};
  const float* q[4] = {qr, qi, qj, qk};
  int t = ob * 4 + ib;
  wt[b_pack_idx(in, out)] = (_Float16)(sgnT[t] * q[srcT[t]][o * 64 + ic]);
}

// --- pack q_rot (f32 row-major) into per-batch B-fragment f16 layout for the agg GEMM ---
__global__ void pack_qb_kernel(const float* __restrict__ q_rot, _Float16* __restrict__ qb) {
  int gid = blockIdx.x * blockDim.x + threadIdx.x;   // B_*N_*D_ = 2^24, row-major coalesced read
  int n = gid & 255;
  int k = (gid >> 8) & (N_ - 1);
  int b = gid >> 21;
  qb[(size_t)b * (N_ * D_) + b_pack_idx(k, n)] = (_Float16)q_rot[gid];
}

// --- top-64 spike*mask per batch (iterative block argmax in LDS, lowest-index tiebreak) ---
__global__ void topk_kernel(const float* __restrict__ spike, const float* __restrict__ mask,
                            const float* __restrict__ time_norm, const int* __restrict__ var_id,
                            int* __restrict__ idx_out, float* __restrict__ tsd, int* __restrict__ vsd) {
  __shared__ float sc[N_];
  __shared__ float rv[256];
  __shared__ int   ri[256];
  int b = blockIdx.x, tid = threadIdx.x;
  for (int n = tid; n < N_; n += 256) sc[n] = spike[b*N_+n] * mask[b*N_+n];
  __syncthreads();
  for (int it = 0; it < KE_; ++it) {
    float bv = -1e30f; int bi = 0x7fffffff;
    for (int n = tid; n < N_; n += 256) { float v = sc[n]; if (v > bv) { bv = v; bi = n; } }
    rv[tid] = bv; ri[tid] = bi;
    __syncthreads();
    for (int s = 128; s > 0; s >>= 1) {
      if (tid < s) {
        float v2 = rv[tid+s]; int i2 = ri[tid+s];
        if (v2 > rv[tid] || (v2 == rv[tid] && i2 < ri[tid])) { rv[tid] = v2; ri[tid] = i2; }
      }
      __syncthreads();
    }
    if (tid == 0) {
      int best = ri[0];
      idx_out[b*KE_+it] = best;
      tsd[b*KE_+it]     = time_norm[b*N_+best];
      vsd[b*KE_+it]     = var_id[b*N_+best];
      sc[best] = -1e30f;
    }
    __syncthreads();
  }
}

// --- h1 = gather(q_rot, idx) @ Wt_seed + b  (one wave per 16x16 tile, K=256) ---
__global__ void seed_gemm_kernel(const float* __restrict__ q_rot, const int* __restrict__ idx,
                                 const _Float16* __restrict__ wt, const float* __restrict__ bias,
                                 float* __restrict__ out) {
  int lane = threadIdx.x;
  int m0 = blockIdx.x * 16;
  int nt = blockIdx.y;
  int mrow = m0 + (lane & 15);
  int b = mrow >> 6, kk = mrow & 63;
  const float* arow = q_rot + ((size_t)b * N_ + idx[b*KE_+kk]) * D_;
  const v16h* wfrag = (const v16h*)wt;
  v8f c = {};
  #pragma unroll
  for (int ks = 0; ks < 8; ++ks) {
    v16h a  = load_a_f32(arow, ks * 32, lane);
    v16h bb = wfrag[(ks * 16 + nt) * 32 + lane];
    c = wmma_f16(a, bb, c);
  }
  int nn = nt * 16 + (lane & 15);
  for (int r = 0; r < 8; ++r) {
    int row = m0 + r + ((lane >> 4) << 3);
    out[(size_t)row * D_ + nn] = c[r] + bias[nn];
  }
}

// --- generic qlin: out = A @ Wt + bias (+ addend), A f32 row-major [*,256] ---
__global__ void qlin_gemm_kernel(const float* __restrict__ A, const _Float16* __restrict__ wt,
                                 const float* __restrict__ bias, const float* __restrict__ addend,
                                 float* __restrict__ out) {
  int lane = threadIdx.x;
  int m0 = blockIdx.x * 16;
  int nt = blockIdx.y;
  const float* arow = A + (size_t)(m0 + (lane & 15)) * D_;
  const v16h* wfrag = (const v16h*)wt;
  v8f c = {};
  #pragma unroll
  for (int ks = 0; ks < 8; ++ks) {
    v16h a  = load_a_f32(arow, ks * 32, lane);
    v16h bb = wfrag[(ks * 16 + nt) * 32 + lane];
    c = wmma_f16(a, bb, c);
  }
  int nn = nt * 16 + (lane & 15);
  for (int r = 0; r < 8; ++r) {
    int row = m0 + r + ((lane >> 4) << 3);
    float v = c[r] + bias[nn];
    if (addend) v += addend[(size_t)row * D_ + nn];
    out[(size_t)row * D_ + nn] = v;
  }
}

// --- incid[b,k,n] f16 + denom[b,k] ---
__global__ void incid_kernel(const float* __restrict__ spike, const float* __restrict__ mask,
                             const float* __restrict__ time_norm, const int* __restrict__ var_id,
                             const float* __restrict__ tsd, const int* __restrict__ vsd,
                             const float* __restrict__ log_dt_aggr,
                             _Float16* __restrict__ incid, float* __restrict__ denom) {
  __shared__ float red[256];
  int k = blockIdx.x, b = blockIdx.y, tid = threadIdx.x;
  float dt = expf(log_dt_aggr[0]);
  dt = fminf(fmaxf(dt, 0.001f), 1.0f);
  float inv = 2.0f / fmaxf(dt, 0.001f);
  float ts = tsd[b*KE_+k];
  int   vs = vsd[b*KE_+k];
  float sum = 0.f;
  size_t base = ((size_t)b * KE_ + k) * N_;
  for (int n = tid; n < N_; n += 256) {
    float delta = fabsf(time_norm[b*N_+n] - ts);
    float v = 0.f;
    if (delta <= dt) {
      float aff = (var_id[b*N_+n] == vs) ? 1.0f : 0.5f;
      v = __expf(-delta * inv) * aff * spike[b*N_+n] * mask[b*N_+n];
    }
    incid[base + n] = (_Float16)v;
    sum += v;
  }
  red[tid] = sum;
  __syncthreads();
  for (int s = 128; s > 0; s >>= 1) {
    if (tid < s) red[tid] += red[tid + s];
    __syncthreads();
  }
  if (tid == 0) denom[b*KE_+k] = fmaxf(red[0], 1e-6f);
}

// --- agg = (incid @ q_rot) / denom, K = 8192, B operand from packed qb ---
__global__ void agg_gemm_kernel(const _Float16* __restrict__ incid, const _Float16* __restrict__ qb,
                                const float* __restrict__ denom, float* __restrict__ agg) {
  int lane = threadIdx.x;
  int nt = blockIdx.x, mt = blockIdx.y, b = blockIdx.z;
  int m0 = mt * 16;
  const _Float16* arow = incid + ((size_t)b * KE_ + m0 + (lane & 15)) * N_;
  const v16h* bfrag = (const v16h*)(qb + (size_t)b * (N_ * D_));
  v8f c = {};
  for (int ks = 0; ks < 256; ++ks) {
    int k0 = ks * 32;
    if (ks + 1 < 256) __builtin_prefetch(arow + k0 + 32, 0, 1);  // global_prefetch_b8
    v16h a  = load_a_f16(arow, k0, lane);
    v16h bb = bfrag[(ks * 16 + nt) * 32 + lane];
    c = wmma_f16(a, bb, c);
  }
  int nn = nt * 16 + (lane & 15);
  for (int r = 0; r < 8; ++r) {
    int row = m0 + r + ((lane >> 4) << 3);
    agg[((size_t)b * KE_ + row) * D_ + nn] = c[r] / denom[b*KE_+row];
  }
}

// --- Mb[b][in->K][hk->N] = scale * sum_d WtQ[in][h*64+d] * K[b,k,h*64+d], packed output ---
// note: wtq here is a plain (unpacked) f16 Wt copy for dot-product reads
__global__ void make_scoremat_kernel(const _Float16* __restrict__ wtq, const float* __restrict__ kmat,
                                     _Float16* __restrict__ mb) {
  int in = blockIdx.x, b = blockIdx.y, hk = threadIdx.x;
  int h = hk >> 6, k = hk & 63;
  const _Float16* wrow = wtq + in * D_ + h * HD_;
  const float*    krow = kmat + ((size_t)b * KE_ + k) * D_ + h * HD_;
  float s = 0.f;
  for (int d = 0; d < HD_; ++d) s += (float)wrow[d] * krow[d];
  mb[(size_t)b * (D_ * D_) + b_pack_idx(in, hk)] = (_Float16)(s * 0.125f);
}

// plain (row-major) Wt copy builder for the two matrices consumed by dot-product kernels
__global__ void build_wt_rowmajor_kernel(const float* __restrict__ qr, const float* __restrict__ qi,
                                         const float* __restrict__ qj, const float* __restrict__ qk,
                                         _Float16* __restrict__ wt) {
  int gid = blockIdx.x * blockDim.x + threadIdx.x;
  int in  = gid >> 8;
  int out = gid & 255;
  int ob = out >> 6, ib = in >> 6, o = out & 63, ic = in & 63;
  const int   srcT[16] = {0,1,2,3, 1,0,3,2, 2,3,0,1, 3,2,1,0};
  const float sgnT[16] = {1.f,-1.f,-1.f,-1.f, 1.f,1.f,-1.f,1.f,
                          1.f,1.f,1.f,-1.f,  1.f,-1.f,1.f,1.f};
  const float* q[4] = {qr, qi, qj, qk};
  int t = ob * 4 + ib;
  wt[in * D_ + out] = (_Float16)(sgnT[t] * q[srcT[t]][o * 64 + ic]);
}

__global__ void make_sbias_kernel(const float* __restrict__ bq, const float* __restrict__ kmat,
                                  float* __restrict__ sbias) {
  int b = blockIdx.x, hk = threadIdx.x;
  int h = hk >> 6, k = hk & 63;
  const float* krow = kmat + ((size_t)b * KE_ + k) * D_ + h * HD_;
  float s = 0.f;
  for (int d = 0; d < HD_; ++d) s += bq[h*HD_+d] * krow[d];
  sbias[b * D_ + hk] = s * 0.125f;
}

// --- VW[b][hk->K][o->N] = sum_d V[b,k,h*64+d] * WtO[h*64+d][o], packed output ---
__global__ void make_vw_kernel(const float* __restrict__ vmat, const _Float16* __restrict__ wto,
                               _Float16* __restrict__ vw) {
  int hk = blockIdx.x, b = blockIdx.y, o = threadIdx.x;
  int h = hk >> 6, k = hk & 63;
  const float* vrow = vmat + ((size_t)b * KE_ + k) * D_ + h * HD_;
  float s = 0.f;
  for (int d = 0; d < HD_; ++d) s += vrow[d] * (float)wto[(h*HD_+d) * D_ + o];
  vw[(size_t)b * (D_ * D_) + b_pack_idx(hk, o)] = (_Float16)s;
}

// --- fused attention per 16-query tile: scores GEMM -> masked softmax -> output GEMM ---
__global__ void fused_attn_kernel(const float* __restrict__ q_rot, const _Float16* __restrict__ mb,
                                  const float* __restrict__ sbias, const _Float16* __restrict__ vw,
                                  const float* __restrict__ bo, const float* __restrict__ time_norm,
                                  const float* __restrict__ mask, const float* __restrict__ tsd,
                                  const float* __restrict__ log_dt_dist, float* __restrict__ out) {
  __shared__ float    s_sc[16 * 256];
  __shared__ _Float16 s_at[16 * 256];
  __shared__ float    s_ts[KE_];
  int lane = threadIdx.x;
  int b = blockIdx.y;
  int n0 = blockIdx.x * 16;
  s_ts[lane]      = tsd[b*KE_ + lane];
  s_ts[lane + 32] = tsd[b*KE_ + lane + 32];

  // A fragments (x tile, K = 256) converted f32->f16 with vector loads, live across n-tiles
  v16h a[8];
  {
    const float* arow = q_rot + ((size_t)b * N_ + n0 + (lane & 15)) * D_;
    #pragma unroll
    for (int ks = 0; ks < 8; ++ks) a[ks] = load_a_f32(arow, ks * 32, lane);
  }
  // phase 1: scores[16][256] = x @ Mb + sbias
  const v16h* mbf = (const v16h*)(mb + (size_t)b * (D_ * D_));
  for (int nt = 0; nt < 16; ++nt) {
    v8f c = {};
    #pragma unroll
    for (int ks = 0; ks < 8; ++ks)
      c = wmma_f16(a[ks], mbf[(ks * 16 + nt) * 32 + lane], c);
    int nn = nt * 16 + (lane & 15);
    for (int r = 0; r < 8; ++r)
      s_sc[(r + ((lane >> 4) << 3)) * 256 + nn] = c[r] + sbias[b * D_ + nn];
  }
  __syncthreads();

  // phase 2: masked softmax; 64 (row,head) pairs -> 2 per lane
  float dtd = expf(log_dt_dist[0]);
  dtd = fminf(fmaxf(dtd, 0.001f), 1.0f);
  for (int pp = 0; pp < 2; ++pp) {
    int p = lane * 2 + pp;
    int row = p >> 2, head = p & 3;
    int n = n0 + row;
    float tn = time_norm[b*N_+n];
    float mk = mask[b*N_+n];
    float*    sp = s_sc + row * 256 + head * 64;
    _Float16* ap = s_at + row * 256 + head * 64;
    float mx = -1e30f;
    for (int k = 0; k < KE_; ++k) {
      float s = sp[k];
      bool ok = (fabsf(tn - s_ts[k]) <= dtd) && (mk != 0.f);
      s = ok ? s : -1e9f;
      sp[k] = s;
      mx = fmaxf(mx, s);
    }
    float sum = 0.f;
    for (int k = 0; k < KE_; ++k) {
      float e = __expf(sp[k] - mx);
      sum += e;
      ap[k] = (_Float16)e;
    }
    float inv = 1.0f / sum;
    for (int k = 0; k < KE_; ++k) ap[k] = (_Float16)((float)ap[k] * inv);
  }
  __syncthreads();

  // phase 3: final = attn @ VW + b_o  (attn A-fragments via 16B LDS vector loads)
  const v16h* vwf = (const v16h*)(vw + (size_t)b * (D_ * D_));
  for (int nt = 0; nt < 16; ++nt) {
    v8f c = {};
    #pragma unroll
    for (int ks = 0; ks < 8; ++ks) {
      v16h aa = load_a_f16(&s_at[(lane & 15) * 256], ks * 32, lane);
      c = wmma_f16(aa, vwf[(ks * 16 + nt) * 32 + lane], c);
    }
    int nn = nt * 16 + (lane & 15);
    for (int r = 0; r < 8; ++r) {
      int mrow = r + ((lane >> 4) << 3);
      out[((size_t)b * N_ + n0 + mrow) * D_ + nn] = c[r] + bo[nn];
    }
  }
}

extern "C" void kernel_launch(void* const* d_in, const int* in_sizes, int n_in,
                              void* d_out, int out_size, void* d_ws, size_t ws_size,
                              hipStream_t stream) {
  const float* q_rot       = (const float*)d_in[0];
  const float* spike       = (const float*)d_in[1];
  const float* time_norm   = (const float*)d_in[2];
  const int*   var_id      = (const int*)d_in[3];
  const float* mask        = (const float*)d_in[4];
  const float* log_dt_aggr = (const float*)d_in[5];
  const float* log_dt_dist = (const float*)d_in[6];
  // qlin params flattened in insertion order: seed, aggr, fc_q, fc_k, fc_v, fc_o; each r,i,j,k,b
  auto qp = [&](int m, int part) { return (const float*)d_in[7 + m * 5 + part]; };

  // workspace carve (256B aligned)
  char* p = (char*)d_ws;
  auto carve = [&](size_t bytes) { char* r = p; p += (bytes + 255) & ~(size_t)255; return r; };
  _Float16* wt[6];   // packed B-fragment layout (seed, aggr, fc_q*, fc_k, fc_v, fc_o*)
  for (int m = 0; m < 6; ++m) wt[m] = (_Float16*)carve((size_t)D_ * D_ * sizeof(_Float16));
  _Float16* wtq_rm = (_Float16*)carve((size_t)D_ * D_ * sizeof(_Float16)); // fc_q row-major
  _Float16* wto_rm = (_Float16*)carve((size_t)D_ * D_ * sizeof(_Float16)); // fc_o row-major
  _Float16* qb    = (_Float16*)carve((size_t)B_ * N_ * D_ * sizeof(_Float16)); // packed q_rot
  int*      idx   = (int*)     carve((size_t)B_ * KE_ * sizeof(int));
  float*    tsd   = (float*)   carve((size_t)B_ * KE_ * sizeof(float));
  int*      vsd   = (int*)     carve((size_t)B_ * KE_ * sizeof(int));
  float*    h_ev  = (float*)   carve((size_t)B_ * KE_ * D_ * sizeof(float));
  _Float16* incid = (_Float16*)carve((size_t)B_ * KE_ * N_ * sizeof(_Float16));
  float*    denom = (float*)   carve((size_t)B_ * KE_ * sizeof(float));
  float*    aggb  = (float*)   carve((size_t)B_ * KE_ * D_ * sizeof(float));
  float*    kmat  = (float*)   carve((size_t)B_ * KE_ * D_ * sizeof(float));
  float*    vmat  = (float*)   carve((size_t)B_ * KE_ * D_ * sizeof(float));
  _Float16* mbW   = (_Float16*)carve((size_t)B_ * D_ * D_ * sizeof(_Float16));
  float*    sbias = (float*)   carve((size_t)B_ * D_ * sizeof(float));
  _Float16* vwW   = (_Float16*)carve((size_t)B_ * D_ * D_ * sizeof(_Float16));

  // 1) Hamilton weights: packed for GEMM consumers; row-major copies for dot-product kernels
  for (int m = 0; m < 6; ++m)
    build_wt_kernel<<<D_, 256, 0, stream>>>(qp(m,0), qp(m,1), qp(m,2), qp(m,3), wt[m]);
  build_wt_rowmajor_kernel<<<D_, 256, 0, stream>>>(qp(2,0), qp(2,1), qp(2,2), qp(2,3), wtq_rm);
  build_wt_rowmajor_kernel<<<D_, 256, 0, stream>>>(qp(5,0), qp(5,1), qp(5,2), qp(5,3), wto_rm);
  // 2) packed f16 q_rot for the aggregation GEMM B operand
  pack_qb_kernel<<<(B_*N_*D_)/256, 256, 0, stream>>>(q_rot, qb);
  // 3) top-64 seeds
  topk_kernel<<<B_, 256, 0, stream>>>(spike, mask, time_norm, var_id, idx, tsd, vsd);
  // 4) h1 = qlin(q_seed, seed)
  seed_gemm_kernel<<<dim3(B_*KE_/16, D_/16), 32, 0, stream>>>(q_rot, idx, wt[0], qp(0,4), h_ev);
  // 5) incid + denom
  incid_kernel<<<dim3(KE_, B_), 256, 0, stream>>>(spike, mask, time_norm, var_id,
                                                  tsd, vsd, log_dt_aggr, incid, denom);
  // 6) agg = incid @ q_rot / denom
  agg_gemm_kernel<<<dim3(D_/16, KE_/16, B_), 32, 0, stream>>>(incid, qb, denom, aggb);
  // 7) h_event = h1 + qlin(agg, aggr)
  qlin_gemm_kernel<<<dim3(B_*KE_/16, D_/16), 32, 0, stream>>>(aggb, wt[1], qp(1,4), h_ev, h_ev);
  // 8) K, V projections
  qlin_gemm_kernel<<<dim3(B_*KE_/16, D_/16), 32, 0, stream>>>(h_ev, wt[3], qp(3,4), nullptr, kmat);
  qlin_gemm_kernel<<<dim3(B_*KE_/16, D_/16), 32, 0, stream>>>(h_ev, wt[4], qp(4,4), nullptr, vmat);
  // 9) per-batch fused matrices: Mb = scale*WqT*K^T (packed), sbias, VW = V*WoT (packed)
  make_scoremat_kernel<<<dim3(D_, B_), 256, 0, stream>>>(wtq_rm, kmat, mbW);
  make_sbias_kernel<<<B_, 256, 0, stream>>>(qp(2,4), kmat, sbias);
  make_vw_kernel<<<dim3(D_, B_), 256, 0, stream>>>(vmat, wto_rm, vwW);
  // 10) fused attention: scores -> masked softmax -> out (+fc_o folded in)
  fused_attn_kernel<<<dim3(N_/16, B_), 32, 0, stream>>>(q_rot, mbW, sbias, vwW, qp(5,4),
                                                        time_norm, mask, tsd, log_dt_dist,
                                                        (float*)d_out);
}
